// CausalSelfAttention_58239756533763
// MI455X (gfx1250) — compile-verified, matrix operand order checked by reading, not run on hardware
//
#include <hip/hip_runtime.h>
#include <hip/hip_bf16.h>

#define B_   4
#define T_   1024
#define M_   1024
#define E_   1024
#define H_   16
#define D_   64
#define HD_  1024
#define J_   2048
#define SCALE_ 0.125f   // 64^-0.5

typedef __attribute__((ext_vector_type(16))) __bf16       bf16x16;
typedef __attribute__((ext_vector_type(8)))  float        f32x8;
typedef __attribute__((ext_vector_type(4)))  float        f32x4;
typedef __attribute__((ext_vector_type(4)))  unsigned int u32x4;
typedef __attribute__((ext_vector_type(4)))  int          i32x4;
typedef __attribute__((ext_vector_type(8)))  int          i32x8;

__device__ __forceinline__ unsigned short f2bf(float f) {
  union { float f; unsigned u; } v; v.f = f;
  unsigned u = v.u;
  u += 0x7FFFu + ((u >> 16) & 1u);     // round-to-nearest-even
  return (unsigned short)(u >> 16);
}

// pack 8 f32 -> 8 bf16 and store as one b128
__device__ __forceinline__ void st_bf8(unsigned short* dst, f32x4 a, f32x4 b) {
  union { unsigned short us[8]; u32x4 v; } pk;
  pk.us[0] = f2bf(a.x); pk.us[1] = f2bf(a.y); pk.us[2] = f2bf(a.z); pk.us[3] = f2bf(a.w);
  pk.us[4] = f2bf(b.x); pk.us[5] = f2bf(b.y); pk.us[6] = f2bf(b.z); pk.us[7] = f2bf(b.w);
  *(u32x4*)dst = pk.v;
}

// ISA 16-bit A/B fragment: halves 0..7 at p[0..7], halves 8..15 at p[16..23]
__device__ __forceinline__ bf16x16 ldfrag(const unsigned short* p) {
  union { u32x4 q[2]; bf16x16 v; } u;
  u.q[0] = *(const u32x4*)p;
  u.q[1] = *(const u32x4*)(p + 16);
  return u.v;
}

__device__ __forceinline__ f32x8 wmma_bf16(bf16x16 a, bf16x16 b, f32x8 c) {
  return __builtin_amdgcn_wmma_f32_16x16x32_bf16(false, a, false, b, (short)0, c, false, false);
}

// ---------------------------------------------------------------------------
// TDM: DMA one (rows x cols) bf16 tile, row stride `stride_elems`, from global
// into LDS at byte offset `lds_off`, inserting 16B of pad after every 128B row
// (=> LDS row pitch 144B = 72 halves, bank-conflict-free for b128 frag reads).
// D# packing per CDNA5 ISA 8.3/8.4; groups 2/3 zero (2D tile).
// This toolchain exposes the 6-arg builtin form.
// ---------------------------------------------------------------------------
__device__ __forceinline__ void tdm_load_tile_bf16(
    unsigned lds_off, const unsigned short* gsrc,
    unsigned rows, unsigned cols, unsigned stride_elems)
{
  const unsigned long long ga = (unsigned long long)(size_t)gsrc;
  u32x4 g0;
  g0[0] = 1u;                                        // count=1 (valid user D#)
  g0[1] = lds_off;                                   // lds_addr
  g0[2] = (unsigned)(ga & 0xFFFFFFFFu);              // global_addr[31:0]
  g0[3] = (unsigned)((ga >> 32) & 0x01FFFFFFu)       // global_addr[56:32]
        | (2u << 30);                                // type=2 (image)
  i32x8 g1;
  g1[0] = (int)((1u << 16)        // data_size = 2 bytes
              | (1u << 20)        // pad_enable
              | (4u << 22)        // pad_interval: 32 DWORDs (=128B row)
              | (3u << 25));      // pad_amount: 4 DWORDs (=16B)
  g1[1] = (int)(cols << 16);      // tensor_dim0[15:0]  @ bits 63:48
  g1[2] = (int)(rows << 16);      // tensor_dim0 hi=0; tensor_dim1[15:0] @ 95:80
  g1[3] = (int)(cols << 16);      // tensor_dim1 hi=0; tile_dim0 @ 127:112
  g1[4] = (int)rows;              // tile_dim1 @ 143:128; tile_dim2=0
  g1[5] = (int)stride_elems;      // tensor_dim0_stride[31:0] @ 191:160
  g1[6] = 0;                      // stride hi, tensor_dim1_stride lo
  g1[7] = 0;
  const i32x4 z4 = {0, 0, 0, 0};
  const i32x8 z8 = {0, 0, 0, 0, 0, 0, 0, 0};
  __builtin_amdgcn_tensor_load_to_lds(g0, g1, z4, z4, z8, 0);
}

// ---------------------------------------------------------------------------
// C = (A @ W + bias) * scale, A:(rows x K) f32, W:(K x N) f32.
// 128x128 block tile, 8 waves, wave tile 64x32, bf16 WMMA 16x16x32.
// Destinations (each optional): f32 `out`, f32 `out2` (kv_to_add_xl region),
// bf16 `outb` (dense, ld = N) feeding the fused attention kernel.
// ---------------------------------------------------------------------------
#define GLDS 40   // LDS stride in halves (80B rows: conflict-free b128 reads)

__global__ __launch_bounds__(256) void gemm_bf16(
    const float* __restrict__ A, const float* __restrict__ W,
    const float* __restrict__ bias, float scale,
    float* __restrict__ out, int ldO,
    float* __restrict__ out2, int ldO2,
    unsigned short* __restrict__ outb,
    int K, int N)
{
  __shared__ unsigned short sA[128 * GLDS];   // [row][k]
  __shared__ unsigned short sB[128 * GLDS];   // [n][k]  (W staged transposed)

  const int tid  = threadIdx.x;
  const int w    = tid >> 5, lane = tid & 31;
  const int lrow = lane & 15, lh = lane >> 4, kb = lh * 8;
  const int wm   = (w >> 2) * 64, wn = (w & 3) * 32;
  const int rowblk = blockIdx.y * 128;
  const int colblk = blockIdx.x * 128;

  f32x8 acc[4][2];
  #pragma unroll
  for (int mi = 0; mi < 4; ++mi)
    #pragma unroll
    for (int ni = 0; ni < 2; ++ni) acc[mi][ni] = (f32x8)0.f;

  const int la_row = tid >> 1, la_cb = (tid & 1) * 16;   // A stage mapping
  const int lb_k   = tid >> 3, lb_n  = (tid & 7) * 16;   // B stage mapping

  for (int kt = 0; kt < K; kt += 32) {
    { // stage A tile 128x32 (f32 -> bf16)
      const float* gp = A + (size_t)(rowblk + la_row) * K + kt + la_cb;
      f32x4 a0 = *(const f32x4*)(gp + 0);
      f32x4 a1 = *(const f32x4*)(gp + 4);
      f32x4 a2 = *(const f32x4*)(gp + 8);
      f32x4 a3 = *(const f32x4*)(gp + 12);
      st_bf8(&sA[la_row * GLDS + la_cb + 0], a0, a1);
      st_bf8(&sA[la_row * GLDS + la_cb + 8], a2, a3);
    }
    { // stage W tile 32x128, transposed into sB[n][k]
      const float* gp = W + (size_t)(kt + lb_k) * N + colblk + lb_n;
      #pragma unroll
      for (int c = 0; c < 16; c += 4) {
        f32x4 f = *(const f32x4*)(gp + c);
        sB[(lb_n + c + 0) * GLDS + lb_k] = f2bf(f.x);
        sB[(lb_n + c + 1) * GLDS + lb_k] = f2bf(f.y);
        sB[(lb_n + c + 2) * GLDS + lb_k] = f2bf(f.z);
        sB[(lb_n + c + 3) * GLDS + lb_k] = f2bf(f.w);
      }
    }
    __syncthreads();

    bf16x16 af[4], bfr[2];
    #pragma unroll
    for (int mi = 0; mi < 4; ++mi)
      af[mi] = ldfrag(&sA[(wm + mi * 16 + lrow) * GLDS + kb]);
    #pragma unroll
    for (int ni = 0; ni < 2; ++ni)
      bfr[ni] = ldfrag(&sB[(wn + ni * 16 + lrow) * GLDS + kb]);
    #pragma unroll
    for (int mi = 0; mi < 4; ++mi)
      #pragma unroll
      for (int ni = 0; ni < 2; ++ni)
        acc[mi][ni] = wmma_bf16(af[mi], bfr[ni], acc[mi][ni]);
    __syncthreads();
  }

  // epilogue: C layout -> row = base + r + 8*lh, col = base + lane%16
  #pragma unroll
  for (int mi = 0; mi < 4; ++mi) {
    #pragma unroll
    for (int ni = 0; ni < 2; ++ni) {
      const int col = colblk + wn + ni * 16 + lrow;
      const float bv = bias ? bias[col] : 0.f;
      #pragma unroll
      for (int r = 0; r < 8; ++r) {
        const int row = rowblk + wm + mi * 16 + r + 8 * lh;
        const float v = (acc[mi][ni][r] + bv) * scale;
        if (out)  out[(size_t)row * ldO + col] = v;
        if (out2) out2[(size_t)row * ldO2 + col] = v;
        if (outb) outb[(size_t)row * N + col] = f2bf(v);
      }
    }
  }
}

// ---------------------------------------------------------------------------
// xl_memory (B,M,2,HD) f32 -> kxl/vxl (B,M,HD) bf16, one pass.
// ---------------------------------------------------------------------------
__global__ __launch_bounds__(256) void xl_to_bf16(
    const float* __restrict__ xl,
    unsigned short* __restrict__ kxl, unsigned short* __restrict__ vxl)
{
  const int gid  = blockIdx.x * 256 + threadIdx.x;  // B*M*2*HD/8 threads
  const int row2 = gid >> 7;                        // (b*M+m)*2+sel
  const int c    = (gid & 127) << 3;
  const float* src = xl + (size_t)row2 * HD_ + c;
  f32x4 a = *(const f32x4*)(src);
  f32x4 b = *(const f32x4*)(src + 4);
  unsigned short* dst = ((row2 & 1) ? vxl : kxl) + (size_t)(row2 >> 1) * HD_ + c;
  st_bf8(dst, a, b);
}

// ---------------------------------------------------------------------------
// Fused flash attention, all-bf16 operands. One block per (b,h,64-query tile),
// 4 waves. K tiles arrive via the Tensor Data Mover (padded into 72-half
// rows); V tiles are staged transposed by the VALU path. Online softmax with
// rel-position bias and causal mask jj <= M + i.
// ---------------------------------------------------------------------------
#define KSTR 72
#define VSTR 72
#define PSTR 72

__global__ __launch_bounds__(128) void attn_fused(
    const unsigned short* __restrict__ qbf,  // (B,T,HD) bf16, already * SCALE
    const unsigned short* __restrict__ kbf,  // (B,T,HD) bf16
    const unsigned short* __restrict__ vbf,  // (B,T,HD) bf16
    const unsigned short* __restrict__ kxl,  // (B,M,HD) bf16
    const unsigned short* __restrict__ vxl,  // (B,M,HD) bf16
    const float* __restrict__ rel,           // (H,T,J) f32
    float* __restrict__ qkv)                 // (B,T,HD) f32
{
  __shared__ unsigned short sK[64 * KSTR];   // [j][d]   (TDM-filled, padded)
  __shared__ unsigned short sV[64 * VSTR];   // [d][j]   (transposed)
  __shared__ unsigned short sP[64 * PSTR];   // [i_local][j]

  const int blk = blockIdx.x;
  const int qt = blk & 15, h = (blk >> 4) & 15, b = blk >> 8;
  const int q0 = qt * 64;
  const int tid = threadIdx.x, w = tid >> 5, lane = tid & 31;
  const int lrow = lane & 15, lh = lane >> 4, kb = lh * 8;
  const unsigned sK_off = (unsigned)(unsigned long long)(size_t)(void*)sK;

  // Q fragments held in registers for the whole j loop (bf16, no conversion)
  bf16x16 qf[2];
  {
    const int qrow = q0 + w * 16 + lrow;
    const unsigned short* qp = qbf + (size_t)(b * T_ + qrow) * HD_ + h * D_;
    #pragma unroll
    for (int kc = 0; kc < 2; ++kc) qf[kc] = ldfrag(qp + kc * 32 + kb);
  }

  float m_i[8], l_i[8];
  f32x8 accO[4];
  #pragma unroll
  for (int r = 0; r < 8; ++r) { m_i[r] = -1e30f; l_i[r] = 0.f; }
  #pragma unroll
  for (int di = 0; di < 4; ++di) accO[di] = (f32x8)0.f;

  // causal bound: largest needed jj = M + q0 + 63
  const int jtiles = (M_ + q0) / 64 + 1;
  const int ld_j  = tid >> 1;          // 0..63
  const int ld_db = (tid & 1) * 32;    // 0 / 32

  for (int jt = 0; jt < jtiles; ++jt) {
    const int jb = jt * 64;
    const unsigned short *kptr, *vptr;
    if (jb < M_) {
      kptr = kxl + (size_t)(b * M_ + jb) * HD_ + h * D_;
      vptr = vxl + (size_t)(b * M_ + jb) * HD_ + h * D_;
    } else {
      const int jc = jb - M_;
      kptr = kbf + (size_t)(b * T_ + jc) * HD_ + h * D_;
      vptr = vbf + (size_t)(b * T_ + jc) * HD_ + h * D_;
    }

    // K tile 64x64 bf16 -> sK via Tensor Data Mover (wave 0 issues the DMA)
    if (w == 0) tdm_load_tile_bf16(sK_off, kptr, 64, 64, HD_);

    if (jt + 1 < jtiles) {                       // global_prefetch_b8
      __builtin_prefetch(vptr + 64 * HD_, 0, 1);
    }
    { // V tile -> sV[d][j] (bf16 transposed scatter)
      const unsigned short* vp = vptr + (size_t)ld_j * HD_ + ld_db;
      #pragma unroll
      for (int c = 0; c < 32; c += 8) {
        union { u32x4 v; unsigned short us[8]; } u;
        u.v = *(const u32x4*)(vp + c);
        #pragma unroll
        for (int i = 0; i < 8; ++i)
          sV[(ld_db + c + i) * VSTR + ld_j] = u.us[i];
      }
    }
    if (w == 0) __builtin_amdgcn_s_wait_tensorcnt(0);
    __syncthreads();

    // S = Q @ K^T : 16 q rows x 64 j cols per wave
    f32x8 sacc[4];
    #pragma unroll
    for (int ni = 0; ni < 4; ++ni) sacc[ni] = (f32x8)0.f;
    #pragma unroll
    for (int kc = 0; kc < 2; ++kc) {
      #pragma unroll
      for (int ni = 0; ni < 4; ++ni) {
        bf16x16 bf = ldfrag(&sK[(ni * 16 + lrow) * KSTR + kc * 32 + kb]);
        sacc[ni] = wmma_bf16(qf[kc], bf, sacc[ni]);
      }
    }

    // rel bias + scale + causal mask + online softmax; P -> sP (bf16)
    #pragma unroll
    for (int r = 0; r < 8; ++r) {
      const int il  = w * 16 + r + 8 * lh;   // local query row 0..63
      const int i_t = q0 + il;               // t index
      const float* relrow = rel + ((size_t)h * T_ + i_t) * J_ + jb;
      const int lim = M_ + i_t;
      float sv[4];
      #pragma unroll
      for (int ni = 0; ni < 4; ++ni) {
        const int jj = jb + ni * 16 + lrow;
        const float s = (sacc[ni][r] + relrow[ni * 16 + lrow]) * SCALE_;
        sv[ni] = (jj > lim) ? -1e30f : s;
      }
      float mx = fmaxf(fmaxf(sv[0], sv[1]), fmaxf(sv[2], sv[3]));
      #pragma unroll
      for (int off = 1; off < 16; off <<= 1)
        mx = fmaxf(mx, __shfl_xor(mx, off, 32));
      const float mnew = fmaxf(m_i[r], mx);
      const float corr = __expf(m_i[r] - mnew);
      float ps = 0.f;
      #pragma unroll
      for (int ni = 0; ni < 4; ++ni) {
        const float p = __expf(sv[ni] - mnew);
        ps += p;
        sP[il * PSTR + ni * 16 + lrow] = f2bf(p);
      }
      #pragma unroll
      for (int off = 1; off < 16; off <<= 1)
        ps += __shfl_xor(ps, off, 32);
      m_i[r] = mnew;
      l_i[r] = l_i[r] * corr + ps;
      #pragma unroll
      for (int di = 0; di < 4; ++di) accO[di][r] *= corr;
    }
    __syncthreads();

    // O += P @ V
    #pragma unroll
    for (int kc = 0; kc < 2; ++kc) {
      bf16x16 pf = ldfrag(&sP[(w * 16 + lrow) * PSTR + kc * 32 + kb]);
      #pragma unroll
      for (int di = 0; di < 4; ++di) {
        bf16x16 vf = ldfrag(&sV[(di * 16 + lrow) * VSTR + kc * 32 + kb]);
        accO[di] = wmma_bf16(pf, vf, accO[di]);
      }
    }
    __syncthreads();
  }

  // normalize and store (B,T,HD)
  #pragma unroll
  for (int di = 0; di < 4; ++di) {
    const int d = di * 16 + lrow;
    #pragma unroll
    for (int r = 0; r < 8; ++r) {
      const int i_t = q0 + w * 16 + r + 8 * lh;
      qkv[(size_t)(b * T_ + i_t) * HD_ + h * D_ + d] = accO[di][r] / l_i[r];
    }
  }
}

// ---------------------------------------------------------------------------
extern "C" void kernel_launch(void* const* d_in, const int* in_sizes, int n_in,
                              void* d_out, int out_size, void* d_ws, size_t ws_size,
                              hipStream_t stream) {
  (void)in_sizes; (void)n_in; (void)out_size; (void)ws_size;
  const float* x   = (const float*)d_in[0];
  const float* xl  = (const float*)d_in[1];
  const float* rel = (const float*)d_in[2];
  const float* Wq  = (const float*)d_in[3];
  const float* bq  = (const float*)d_in[4];
  const float* Wk  = (const float*)d_in[5];
  const float* bk  = (const float*)d_in[6];
  const float* Wv  = (const float*)d_in[7];
  const float* bv  = (const float*)d_in[8];
  const float* Wo  = (const float*)d_in[9];
  const float* bo  = (const float*)d_in[10];

  float* out    = (float*)d_out;
  float* out_kv = out + (size_t)B_ * T_ * E_;   // (B,M,2,HD) f32

  const size_t BTHD = (size_t)B_ * T_ * HD_;
  float*          oacc = (float*)d_ws;                          // (B,T,HD) f32
  unsigned short* q_bf = (unsigned short*)(oacc + BTHD);        // (B,T,HD) bf16
  unsigned short* k_bf = q_bf + BTHD;
  unsigned short* v_bf = k_bf + BTHD;
  unsigned short* kxl  = v_bf + BTHD;                           // (B,M,HD) bf16
  unsigned short* vxl  = kxl + (size_t)B_ * M_ * HD_;

  dim3 blk(256);
  dim3 grd(HD_ / 128, (B_ * T_) / 128);

  // xl_memory -> bf16 (one pass)
  xl_to_bf16<<<dim3((B_ * M_ * 2 * HD_ / 8) / 256), blk, 0, stream>>>(xl, kxl, vxl);

  // Q/K/V projections: bf16 mirrors for attention; K,V f32 stream straight
  // into the kv_to_add_xl output region (stride 2*HD).
  gemm_bf16<<<grd, blk, 0, stream>>>(x, Wq, bq, SCALE_, nullptr, 0, nullptr,      0,       q_bf, E_, HD_);
  gemm_bf16<<<grd, blk, 0, stream>>>(x, Wk, bk, 1.0f,   nullptr, 0, out_kv,       2 * HD_, k_bf, E_, HD_);
  gemm_bf16<<<grd, blk, 0, stream>>>(x, Wv, bv, 1.0f,   nullptr, 0, out_kv + HD_, 2 * HD_, v_bf, E_, HD_);

  attn_fused<<<dim3(B_ * H_ * (T_ / 64)), dim3(128), 0, stream>>>(
      q_bf, k_bf, v_bf, kxl, vxl, rel, oacc);

  // output projection -> d_out
  gemm_bf16<<<dim3(E_ / 128, (B_ * T_) / 128), blk, 0, stream>>>(
      oacc, Wo, bo, 1.0f, out, E_, nullptr, 0, nullptr, HD_, E_);
}
